// tree_gru_onehot_s2s_60971355734176
// MI455X (gfx1250) — matrix operand order repaired and checked by reading, not compiled, not run on hardware
//
#include <hip/hip_runtime.h>
#include <hip/hip_bf16.h>
#include <math.h>

// ---------------- fixed model dims (from reference) ----------------
#define HDIM   128
#define HEADS  4
#define CONVS  3
#define TSEQ   (CONVS + 1)
#define EPSV   1e-5f
#define G3     (3 * HDIM)   // 384 GRU gate width

typedef __attribute__((ext_vector_type(16))) _Float16 v16h;
typedef __attribute__((ext_vector_type(8)))  float    v8f;

// ---------------- wave32 reduction ----------------
__device__ __forceinline__ float wave_red_add(float x) {
#pragma unroll
  for (int off = 16; off > 0; off >>= 1) x += __shfl_xor(x, off, 32);
  return x;
}

__device__ __forceinline__ void atomicMaxFloat(float* addr, float val) {
  // sign-split trick: int ordering matches float for >=0, reversed (as uint) for <0
  if (!(val < 0.0f)) atomicMax((int*)addr, __float_as_int(val));
  else               atomicMin((unsigned int*)addr, (unsigned int)__float_as_int(val));
}

// =====================================================================
// Generic WMMA GEMM:  C[M,Nn] = A[M,K] (f32) * Bt[Nn,K] (f16)^T
//                     (+ bias[Nn]) (+= C if accumulate) (+ addmat)
// One wave computes a 16x64 tile of C: 4 accumulators share one A
// fragment.  Per k-step, ALL 4 B fragments are loaded first and a
// sched_barrier(0) pins them before the WMMA group, so the loads
// clause together (one s_wait per k-step) and the 4 independent WMMAs
// issue back-to-back.  Live set ~110 VGPRs -> no spill, decent
// occupancy.  A re-reads (2x for Nn=128) are absorbed by the 192 MB L2.
// Requirements (all call sites satisfy them): Nn % 64 == 0, K % 32 == 0.
// A fragment: lane holds row (lane&15), 16 contiguous K values (same order as B):
//   lanes 0-15 : row M=lane,    K = k0..k0+15
//   lanes 16-31: row M=lane-16, K = k0+16..k0+31
// B fragment: lane holds column (lane&15), 16 contiguous K values
//   (lanes<16 -> k0..k0+15, lanes>=16 -> k0+16..k0+31).
// =====================================================================
#define NSUB 4
__global__ void k_wmma_gemm(const float* __restrict__ A, int lda,
                            const _Float16* __restrict__ Bt, int ldb,
                            const float* __restrict__ bias,
                            const float* __restrict__ addmat,
                            float* __restrict__ C, int ldc,
                            int M, int Nn, int K, int accumulate) {
  const int lane  = threadIdx.x & 31;
  const int l15   = lane & 15;
  const bool hi   = lane >= 16;
  const int mBase = blockIdx.x * 16;
  const int nBase = blockIdx.y * (16 * NSUB);
  int arow = mBase + l15; if (arow >= M) arow = M - 1;

  v8f acc[NSUB] = {};
  if (accumulate) {
#pragma unroll
    for (int s = 0; s < NSUB; ++s) {
      const int col = nBase + s * 16 + l15;
#pragma unroll
      for (int r = 0; r < 8; ++r) {
        int mr = mBase + r + (hi ? 8 : 0);
        if (mr < M) acc[s][r] = C[(size_t)mr * ldc + col];
      }
    }
  }

  for (int k0 = 0; k0 < K; k0 += 32) {
    const int ks = k0 + (hi ? 16 : 0);
    const float* ap = A + (size_t)arow * lda + ks;
    v16h af;
#pragma unroll
    for (int i = 0; i < 8; ++i) {
      af[i]     = (_Float16)ap[i];
      af[i + 8] = (_Float16)ap[i + 8];
    }
    // stage ALL B fragments for this k-step...
    v16h bf[NSUB];
#pragma unroll
    for (int s = 0; s < NSUB; ++s) {
      const int col = nBase + s * 16 + l15;
      const _Float16* bp = Bt + (size_t)col * ldb + k0 + (hi ? 16 : 0);
#pragma unroll
      for (int i = 0; i < 16; ++i) bf[s][i] = bp[i];
    }
    // ...and pin them: nothing may cross, so all bf[] stay live in
    // distinct registers and the WMMAs below issue back-to-back.
    __builtin_amdgcn_sched_barrier(0);
#pragma unroll
    for (int s = 0; s < NSUB; ++s) {
      acc[s] = __builtin_amdgcn_wmma_f32_16x16x32_f16(false, af, false, bf[s],
                                                      (short)0, acc[s], false, false);
    }
  }

#pragma unroll
  for (int s = 0; s < NSUB; ++s) {
    const int col = nBase + s * 16 + l15;
    const float bv = bias ? bias[col] : 0.0f;
#pragma unroll
    for (int r = 0; r < 8; ++r) {
      int mr = mBase + r + (hi ? 8 : 0);
      if (mr < M) {
        float v = acc[s][r] + bv;
        if (addmat) v += addmat[(size_t)mr * ldc + col];
        C[(size_t)mr * ldc + col] = v;
      }
    }
  }
}

// ---------------- weight prep ----------------
__global__ void k_transpose_f16(const float* __restrict__ W, _Float16* __restrict__ Wt,
                                int K, int Nn) {           // W[K,Nn] -> Wt[Nn,K]
  size_t idx = blockIdx.x * (size_t)blockDim.x + threadIdx.x;
  size_t total = (size_t)K * Nn;
  if (idx >= total) return;
  int k = (int)(idx / Nn), n = (int)(idx % Nn);
  Wt[(size_t)n * K + k] = (_Float16)W[idx];
}

__global__ void k_convert_f16(const float* __restrict__ W, _Float16* __restrict__ Wt, size_t n) {
  size_t idx = blockIdx.x * (size_t)blockDim.x + threadIdx.x;
  if (idx < n) Wt[idx] = (_Float16)W[idx];
}

// ---------------- utility ----------------
__global__ void k_fill(float* p, float v, size_t n) {
  size_t i = blockIdx.x * (size_t)blockDim.x + threadIdx.x;
  if (i < n) p[i] = v;
}

__global__ void k_gather_rows(const int* __restrict__ wid, const float* __restrict__ table,
                              float* __restrict__ h, int n) {
  size_t idx = blockIdx.x * (size_t)blockDim.x + threadIdx.x;
  size_t total = (size_t)n * (HDIM / 4);
  if (idx >= total) return;
  size_t row = idx / (HDIM / 4), c4 = idx % (HDIM / 4);
  const float4* t4 = (const float4*)table;
  float4* h4 = (float4*)h;
  h4[row * (HDIM / 4) + c4] = t4[(size_t)wid[row] * (HDIM / 4) + c4];
}

__global__ void k_count(const int* __restrict__ gid, float* __restrict__ counts, int n) {
  int i = blockIdx.x * blockDim.x + threadIdx.x;
  if (i < n) atomicAdd(&counts[gid[i]], 1.0f);
}

__global__ void k_meanacc(const float* __restrict__ h, const int* __restrict__ gid,
                          float* __restrict__ slot, int n) {
  size_t idx = blockIdx.x * (size_t)blockDim.x + threadIdx.x;
  if (idx >= (size_t)n * HDIM) return;
  int row = (int)(idx / HDIM), c = (int)(idx & (HDIM - 1));
  atomicAdd(&slot[(size_t)gid[row] * HDIM + c], h[idx]);
}

__global__ void k_divcounts(float* __restrict__ slot, const float* __restrict__ counts, int g) {
  size_t idx = blockIdx.x * (size_t)blockDim.x + threadIdx.x;
  if (idx >= (size_t)g * HDIM) return;
  slot[idx] /= fmaxf(counts[idx / HDIM], 1.0f);
}

__global__ void k_axpy(float* __restrict__ dst, const float* __restrict__ src, size_t n) {
  size_t i = blockIdx.x * (size_t)blockDim.x + threadIdx.x;
  if (i < n) dst[i] += src[i];
}

__global__ void k_scale_out(const float* __restrict__ src, float* __restrict__ dst,
                            float s, size_t n) {
  size_t i = blockIdx.x * (size_t)blockDim.x + threadIdx.x;
  if (i < n) dst[i] = src[i] * s;
}

// ---------------- GAT attention ----------------
__global__ void k_alpha(const float* __restrict__ z, const float* __restrict__ a,
                        float* __restrict__ as, float* __restrict__ ad, int n) {
  int wave = (int)((blockIdx.x * (size_t)blockDim.x + threadIdx.x) >> 5);
  int lane = threadIdx.x & 31;
  if (wave >= n) return;
  float ps = 0.0f, pd = 0.0f;
#pragma unroll
  for (int i = 0; i < HDIM / 32; ++i) {
    int c = lane + i * 32;
    float zv = z[(size_t)wave * HDIM + c];
    ps += zv * a[c];
    pd += zv * a[HDIM + c];
  }
  ps = wave_red_add(ps); pd = wave_red_add(pd);
  if (lane == 0) { as[wave] = ps; ad[wave] = pd; }
}

__global__ void k_edge_e_max(const int* __restrict__ src, const int* __restrict__ dst,
                             const float* __restrict__ as, const float* __restrict__ ad,
                             float* __restrict__ ebuf, float* __restrict__ m, int E) {
  int i = blockIdx.x * blockDim.x + threadIdx.x;
  if (i >= E) return;
  float e = as[src[i]] + ad[dst[i]];
  e = (e > 0.0f) ? e : 0.01f * e;     // leaky_relu 0.01
  ebuf[i] = e;
  atomicMaxFloat(&m[dst[i]], e);
}

__global__ void k_edge_exp_den(const int* __restrict__ dst, float* __restrict__ ebuf,
                               const float* __restrict__ m, float* __restrict__ den, int E) {
  int i = blockIdx.x * blockDim.x + threadIdx.x;
  if (i >= E) return;
  float ex = expf(ebuf[i] - m[dst[i]]);
  ebuf[i] = ex;
  atomicAdd(&den[dst[i]], ex);
}

__global__ void k_edge_agg(const int* __restrict__ src, const int* __restrict__ dst,
                           const float* __restrict__ ebuf, const float* __restrict__ den,
                           const float* __restrict__ z, float* __restrict__ agg) {
  int e = blockIdx.x;
  int c = threadIdx.x;
  int s = src[e], d = dst[e];
  float w = ebuf[e] / den[d];
  atomicAdd(&agg[(size_t)d * HDIM + c], w * z[(size_t)s * HDIM + c]);
}

// ---------------- BatchNorm (training-mode, biased var) ----------------
__global__ void k_bn_stats(float* __restrict__ agg, float* __restrict__ sums, int n) {
  __shared__ float rs[256], rq[256];
  int c = threadIdx.x & (HDIM - 1);
  int half = threadIdx.x >> 7;
  int r0 = blockIdx.x * 512;
  int rend = r0 + 512; if (rend > n) rend = n;
  float s = 0.0f, q = 0.0f;
  for (int r = r0 + half; r < rend; r += 2) {
    size_t i = (size_t)r * HDIM + c;
    float v = fmaxf(agg[i], 0.0f);      // relu in place
    agg[i] = v;
    s += v; q += v * v;
  }
  rs[threadIdx.x] = s; rq[threadIdx.x] = q;
  __syncthreads();
  if (half == 0) {
    atomicAdd(&sums[c],        rs[c] + rs[c + HDIM]);
    atomicAdd(&sums[HDIM + c], rq[c] + rq[c + HDIM]);
  }
}

__global__ void k_bn_finalize(float* __restrict__ sums, int n) {
  int c = threadIdx.x;
  if (c >= HDIM) return;
  float mu  = sums[c] / (float)n;
  float var = sums[HDIM + c] / (float)n - mu * mu;
  sums[c] = mu;
  sums[HDIM + c] = rsqrtf(var + EPSV);
}

__global__ void k_bn_apply(const float* __restrict__ agg, const float* __restrict__ sums,
                           const float* __restrict__ g, const float* __restrict__ b,
                           float* __restrict__ zo, int n) {
  size_t idx = blockIdx.x * (size_t)blockDim.x + threadIdx.x;
  if (idx >= (size_t)n * HDIM) return;
  int c = (int)(idx & (HDIM - 1));
  zo[idx] = (agg[idx] - sums[c]) * sums[HDIM + c] * g[c] + b[c];
}

// ---------------- out_b + residual + LayerNorm (one wave per row) ----------------
__global__ void k_ln_res(const float* __restrict__ y, const float* __restrict__ ob,
                         const float* __restrict__ lg, const float* __restrict__ lb,
                         float* __restrict__ h, int n) {
  int wave = (int)((blockIdx.x * (size_t)blockDim.x + threadIdx.x) >> 5);
  int lane = threadIdx.x & 31;
  if (wave >= n) return;
  size_t base = (size_t)wave * HDIM;
  float v[4]; float s = 0.0f, q = 0.0f;
#pragma unroll
  for (int i = 0; i < 4; ++i) {
    int c = lane + i * 32;
    v[i] = y[base + c] + ob[c] + h[base + c];
    s += v[i]; q += v[i] * v[i];
  }
  s = wave_red_add(s); q = wave_red_add(q);
  float mu = s * (1.0f / HDIM);
  float var = q * (1.0f / HDIM) - mu * mu;
  float istd = rsqrtf(var + EPSV);
#pragma unroll
  for (int i = 0; i < 4; ++i) {
    int c = lane + i * 32;
    h[base + c] = (v[i] - mu) * istd * lg[c] + lb[c];
  }
}

// ---------------- GRU cell pointwise ----------------
__global__ void k_gru_point(const float* __restrict__ gi, const float* __restrict__ gh,
                            float* __restrict__ hb, float* __restrict__ oout,
                            int ostride, int g) {
  size_t idx = blockIdx.x * (size_t)blockDim.x + threadIdx.x;
  if (idx >= (size_t)g * HDIM) return;
  int gg = (int)(idx >> 7);
  int c  = (int)(idx & (HDIM - 1));
  size_t b = (size_t)gg * G3;
  float r  = 1.0f / (1.0f + expf(-(gi[b + c] + gh[b + c])));
  float zt = 1.0f / (1.0f + expf(-(gi[b + HDIM + c] + gh[b + HDIM + c])));
  float nn = tanhf(gi[b + 2 * HDIM + c] + r * gh[b + 2 * HDIM + c]);
  float hp = hb[idx];
  float hn = (1.0f - zt) * nn + zt * hp;
  hb[idx] = hn;
  if (oout) oout[(size_t)gg * ostride + c] = hn;
}

// =====================================================================
// Host orchestration
// =====================================================================
static inline unsigned cdiv(size_t n, int b) { return (unsigned)((n + b - 1) / b); }

extern "C" void kernel_launch(void* const* d_in, const int* in_sizes, int n_in,
                              void* d_out, int out_size, void* d_ws, size_t ws_size,
                              hipStream_t stream) {
  // ---- inputs (setup_inputs dict order) ----
  const int*   wid    = (const int*)d_in[0];
  const int*   src    = (const int*)d_in[1];
  const int*   dst    = (const int*)d_in[2];
  const int*   gid    = (const int*)d_in[3];
  const float* emb    = (const float*)d_in[5];
  const float* projW  = (const float*)d_in[6];
  const float* projB  = (const float*)d_in[7];
  const float* gatW   = (const float*)d_in[8];
  const float* gatA   = (const float*)d_in[9];
  const float* bnG    = (const float*)d_in[10];
  const float* bnB    = (const float*)d_in[11];
  const float* outW   = (const float*)d_in[12];
  const float* outB   = (const float*)d_in[13];
  const float* lnG    = (const float*)d_in[14];
  const float* lnB    = (const float*)d_in[15];
  const float* wih0   = (const float*)d_in[16];
  const float* whh0   = (const float*)d_in[17];
  const float* bih0   = (const float*)d_in[18];
  const float* bhh0   = (const float*)d_in[19];
  const float* wih1   = (const float*)d_in[20];
  const float* whh1   = (const float*)d_in[21];
  const float* bih1   = (const float*)d_in[22];
  const float* bhh1   = (const float*)d_in[23];

  const int N = in_sizes[0];
  const int E = in_sizes[1];
  const int V = in_sizes[5] / HDIM;
  const int G = out_size / HDIM;

  // ---- workspace bump allocator ----
  char* cur = (char*)d_ws;
  auto alloc = [&](size_t bytes) -> void* {
    void* r = (void*)cur;
    cur += (bytes + 255) & ~(size_t)255;
    return r;
  };
  float* table  = (float*)alloc((size_t)V * HDIM * 4);
  float* h      = (float*)alloc((size_t)N * HDIM * 4);
  float* z      = (float*)alloc((size_t)N * HDIM * 4);
  float* agg    = (float*)alloc((size_t)N * HDIM * 4);
  float* y      = (float*)alloc((size_t)N * HDIM * 4);
  float* asrc   = (float*)alloc((size_t)N * 4);
  float* adst   = (float*)alloc((size_t)N * 4);
  float* mbuf   = (float*)alloc((size_t)N * 4);
  float* den    = (float*)alloc((size_t)N * 4);
  float* ebuf   = (float*)alloc((size_t)E * 4);
  float* bnsums = (float*)alloc(2 * HDIM * 4);
  float* counts = (float*)alloc((size_t)G * 4);
  float* seq    = (float*)alloc((size_t)TSEQ * G * HDIM * 4);
  float* o1     = (float*)alloc((size_t)TSEQ * G * 2 * HDIM * 4);
  float* hbuf   = (float*)alloc((size_t)G * HDIM * 4);
  float* hsum   = (float*)alloc((size_t)G * HDIM * 4);
  float* gibuf  = (float*)alloc((size_t)G * G3 * 4);
  float* ghbuf  = (float*)alloc((size_t)G * G3 * 4);
  _Float16* p2t   = (_Float16*)alloc((size_t)HDIM * HDIM * 2);
  _Float16* gatwt = (_Float16*)alloc((size_t)CONVS * HEADS * HDIM * HDIM * 2);
  _Float16* outwt = (_Float16*)alloc((size_t)CONVS * HDIM * (HEADS * HDIM) * 2);
  _Float16* wih0h = (_Float16*)alloc((size_t)2 * G3 * HDIM * 2);
  _Float16* whh0h = (_Float16*)alloc((size_t)2 * G3 * HDIM * 2);
  _Float16* wih1h = (_Float16*)alloc((size_t)2 * G3 * 2 * HDIM * 2);
  _Float16* whh1h = (_Float16*)alloc((size_t)2 * G3 * HDIM * 2);
  if ((size_t)(cur - (char*)d_ws) > ws_size) return;  // workspace too small

  const int BLK = 256;

  // ---- weight prep: f16 transposes/conversions ----
  k_transpose_f16<<<cdiv((size_t)HDIM * HDIM, BLK), BLK, 0, stream>>>(projW + (size_t)V * HDIM, p2t, HDIM, HDIM);
  for (int j = 0; j < CONVS; ++j)
    for (int i = 0; i < HEADS; ++i) {
      size_t off = ((size_t)j * HEADS + i) * HDIM * HDIM;
      k_transpose_f16<<<cdiv((size_t)HDIM * HDIM, BLK), BLK, 0, stream>>>(gatW + off, gatwt + off, HDIM, HDIM);
    }
  for (int j = 0; j < CONVS; ++j)
    k_transpose_f16<<<cdiv((size_t)HEADS * HDIM * HDIM, BLK), BLK, 0, stream>>>(
        outW + (size_t)j * HEADS * HDIM * HDIM, outwt + (size_t)j * HDIM * HEADS * HDIM,
        HEADS * HDIM, HDIM);
  k_convert_f16<<<cdiv((size_t)2 * G3 * HDIM, BLK), BLK, 0, stream>>>(wih0, wih0h, (size_t)2 * G3 * HDIM);
  k_convert_f16<<<cdiv((size_t)2 * G3 * HDIM, BLK), BLK, 0, stream>>>(whh0, whh0h, (size_t)2 * G3 * HDIM);
  k_convert_f16<<<cdiv((size_t)2 * G3 * 2 * HDIM, BLK), BLK, 0, stream>>>(wih1, wih1h, (size_t)2 * G3 * 2 * HDIM);
  k_convert_f16<<<cdiv((size_t)2 * G3 * HDIM, BLK), BLK, 0, stream>>>(whh1, whh1h, (size_t)2 * G3 * HDIM);

  // ---- vocab table: table = projW[:V] + emb @ projW[V:] + projB, then gather ----
  k_wmma_gemm<<<dim3(V / 16, HDIM / 64), 32, 0, stream>>>(emb, HDIM, p2t, HDIM, projB, projW,
                                                          table, HDIM, V, HDIM, HDIM, 0);
  k_gather_rows<<<cdiv((size_t)N * (HDIM / 4), BLK), BLK, 0, stream>>>(wid, table, h, N);

  // ---- per-graph counts + seq[0] = mean_nodes(h) ----
  k_fill<<<cdiv((size_t)G, BLK), BLK, 0, stream>>>(counts, 0.0f, (size_t)G);
  k_count<<<cdiv((size_t)N, BLK), BLK, 0, stream>>>(gid, counts, N);
  k_fill<<<cdiv((size_t)G * HDIM, BLK), BLK, 0, stream>>>(seq, 0.0f, (size_t)G * HDIM);
  k_meanacc<<<cdiv((size_t)N * HDIM, BLK), BLK, 0, stream>>>(h, gid, seq, N);
  k_divcounts<<<cdiv((size_t)G * HDIM, BLK), BLK, 0, stream>>>(seq, counts, G);

  // ---- GAT convolutions ----
  const unsigned mtiles = (unsigned)((N + 15) / 16);
  for (int j = 0; j < CONVS; ++j) {
    for (int i = 0; i < HEADS; ++i) {
      const size_t hoff = ((size_t)j * HEADS + i);
      // z = h @ gat_W[j,i]
      k_wmma_gemm<<<dim3(mtiles, HDIM / 64), 32, 0, stream>>>(
          h, HDIM, gatwt + hoff * HDIM * HDIM, HDIM, nullptr, nullptr, z, HDIM, N, HDIM, HDIM, 0);
      // per-node attention logits
      k_alpha<<<cdiv((size_t)N * 32, 128), 128, 0, stream>>>(z, gatA + hoff * 2 * HDIM, asrc, adst, N);
      // edge softmax + aggregate
      k_fill<<<cdiv((size_t)N, BLK), BLK, 0, stream>>>(mbuf, -INFINITY, (size_t)N);
      k_fill<<<cdiv((size_t)N, BLK), BLK, 0, stream>>>(den, 0.0f, (size_t)N);
      k_fill<<<cdiv((size_t)N * HDIM, BLK), BLK, 0, stream>>>(agg, 0.0f, (size_t)N * HDIM);
      k_edge_e_max<<<cdiv((size_t)E, BLK), BLK, 0, stream>>>(src, dst, asrc, adst, ebuf, mbuf, E);
      k_edge_exp_den<<<cdiv((size_t)E, BLK), BLK, 0, stream>>>(dst, ebuf, mbuf, den, E);
      k_edge_agg<<<(unsigned)E, HDIM, 0, stream>>>(src, dst, ebuf, den, z, agg);
      // relu + batchnorm -> z (reused as normalized head output)
      k_fill<<<1, BLK, 0, stream>>>(bnsums, 0.0f, (size_t)2 * HDIM);
      k_bn_stats<<<cdiv((size_t)N, 512), 256, 0, stream>>>(agg, bnsums, N);
      k_bn_finalize<<<1, HDIM, 0, stream>>>(bnsums, N);
      k_bn_apply<<<cdiv((size_t)N * HDIM, BLK), BLK, 0, stream>>>(
          agg, bnsums, bnG + hoff * HDIM, bnB + hoff * HDIM, z, N);
      // y (+)= head @ out_W chunk   (out_b is applied once, later, in k_ln_res)
      k_wmma_gemm<<<dim3(mtiles, HDIM / 64), 32, 0, stream>>>(
          z, HDIM, outwt + (size_t)j * HDIM * HEADS * HDIM + (size_t)i * HDIM, HEADS * HDIM,
          nullptr, nullptr, y, HDIM, N, HDIM, HDIM, (i == 0) ? 0 : 1);
    }
    // y + out_b + residual(h) -> layernorm -> h
    k_ln_res<<<cdiv((size_t)N * 32, 128), 128, 0, stream>>>(
        y, outB + (size_t)j * HDIM, lnG + (size_t)j * HDIM, lnB + (size_t)j * HDIM, h, N);
    // seq[j+1] = mean_nodes(h)
    float* slot = seq + (size_t)(j + 1) * G * HDIM;
    k_fill<<<cdiv((size_t)G * HDIM, BLK), BLK, 0, stream>>>(slot, 0.0f, (size_t)G * HDIM);
    k_meanacc<<<cdiv((size_t)N * HDIM, BLK), BLK, 0, stream>>>(h, gid, slot, N);
    k_divcounts<<<cdiv((size_t)G * HDIM, BLK), BLK, 0, stream>>>(slot, counts, G);
  }

  // ---- bi-GRU, layer 0 (input seq [T][G][H]) then layer 1 (input o1 [T][G][2H]) ----
  k_fill<<<cdiv((size_t)G * HDIM, BLK), BLK, 0, stream>>>(hsum, 0.0f, (size_t)G * HDIM);
  for (int dir = 0; dir < 2; ++dir) {
    k_fill<<<cdiv((size_t)G * HDIM, BLK), BLK, 0, stream>>>(hbuf, 0.0f, (size_t)G * HDIM);
    for (int s = 0; s < TSEQ; ++s) {
      int t = dir ? (TSEQ - 1 - s) : s;
      k_wmma_gemm<<<dim3(G / 16, G3 / 64), 32, 0, stream>>>(
          seq + (size_t)t * G * HDIM, HDIM, wih0h + (size_t)dir * G3 * HDIM, HDIM,
          bih0 + (size_t)dir * G3, nullptr, gibuf, G3, G, G3, HDIM, 0);
      k_wmma_gemm<<<dim3(G / 16, G3 / 64), 32, 0, stream>>>(
          hbuf, HDIM, whh0h + (size_t)dir * G3 * HDIM, HDIM,
          bhh0 + (size_t)dir * G3, nullptr, ghbuf, G3, G, G3, HDIM, 0);
      k_gru_point<<<cdiv((size_t)G * HDIM, BLK), BLK, 0, stream>>>(
          gibuf, ghbuf, hbuf, o1 + (size_t)t * G * 2 * HDIM + (size_t)dir * HDIM, 2 * HDIM, G);
    }
    k_axpy<<<cdiv((size_t)G * HDIM, BLK), BLK, 0, stream>>>(hsum, hbuf, (size_t)G * HDIM);
  }
  for (int dir = 0; dir < 2; ++dir) {
    k_fill<<<cdiv((size_t)G * HDIM, BLK), BLK, 0, stream>>>(hbuf, 0.0f, (size_t)G * HDIM);
    for (int s = 0; s < TSEQ; ++s) {
      int t = dir ? (TSEQ - 1 - s) : s;
      k_wmma_gemm<<<dim3(G / 16, G3 / 64), 32, 0, stream>>>(
          o1 + (size_t)t * G * 2 * HDIM, 2 * HDIM, wih1h + (size_t)dir * G3 * 2 * HDIM, 2 * HDIM,
          bih1 + (size_t)dir * G3, nullptr, gibuf, G3, G, G3, 2 * HDIM, 0);
      k_wmma_gemm<<<dim3(G / 16, G3 / 64), 32, 0, stream>>>(
          hbuf, HDIM, whh1h + (size_t)dir * G3 * HDIM, HDIM,
          bhh1 + (size_t)dir * G3, nullptr, ghbuf, G3, G, G3, HDIM, 0);
      k_gru_point<<<cdiv((size_t)G * HDIM, BLK), BLK, 0, stream>>>(
          gibuf, ghbuf, hbuf, nullptr, 0, G);
    }
    k_axpy<<<cdiv((size_t)G * HDIM, BLK), BLK, 0, stream>>>(hsum, hbuf, (size_t)G * HDIM);
  }

  // ---- output: mean of the 4 final hidden states ----
  k_scale_out<<<cdiv((size_t)G * HDIM, BLK), BLK, 0, stream>>>(hsum, (float*)d_out, 0.25f, (size_t)G * HDIM);
}
